// MHSA_36129264894513
// MI455X (gfx1250) — matrix operand order repaired and checked by reading, not compiled, not run on hardware
//
#include <hip/hip_runtime.h>
#include <hip/hip_bf16.h>

typedef _Float16 h16;
typedef __attribute__((ext_vector_type(8)))  _Float16 v8h;
typedef __attribute__((ext_vector_type(16))) _Float16 v16h;
typedef __attribute__((ext_vector_type(8)))  float    v8f;

#define B_  4
#define N_  2048
#define C_  1024
#define H_  16
#define HD_ 64
#define M_  (B_ * N_)   // 8192
#define K3_ (3 * C_)    // 3072

static __device__ __forceinline__ v16h cat8(v8h lo, v8h hi) {
  v16h r;
#pragma unroll
  for (int i = 0; i < 8; ++i) { r[i] = lo[i]; r[i + 8] = hi[i]; }
  return r;
}

// Async global->LDS copy of 16 bytes per lane (CDNA5 ASYNCcnt path).
static __device__ __forceinline__ void async_ld16(void* lds, const void* g) {
  unsigned l = (unsigned)(uintptr_t)lds;
  asm volatile("global_load_async_to_lds_b128 %0, %1, off" :: "v"(l), "v"(g) : "memory");
}
static __device__ __forceinline__ void wait_async0() {
  asm volatile("s_wait_asynccnt 0" ::: "memory");
}

// Load a 32x16 B fragment from a k-major LDS tile (lane&15 = n column; k: koff..koff+7, koff+16..koff+23, koff = (lane>=16)*8).
static __device__ __forceinline__ v16h ldsb(const h16* col, int ld, int lane) {
  const int koff = (lane >> 4) << 3;
  v16h r;
#pragma unroll
  for (int i = 0; i < 8; ++i) { r[i] = col[(koff + i) * ld]; r[i + 8] = col[(16 + koff + i) * ld]; }
  return r;
}

// A fragment 16x32 f16. base -> element (row0, k0) of row-major [*, ld].
// lane = M row (lane&15); halves 0..7 = K {koff..koff+7},
// halves 8..15 = K {koff+16..koff+23}, koff = (lane>=16)*8.
static __device__ __forceinline__ v16h load_afrag(const h16* base, int ld, int lane) {
  const int row  = lane & 15;
  const int koff = (lane >> 4) << 3;
  const h16* p = base + (size_t)row * ld + koff;
  return cat8(*(const v8h*)(p), *(const v8h*)(p + 16));
}

static __device__ __forceinline__ v8f wmma_f16(v16h a, v16h b, v8f c) {
  return __builtin_amdgcn_wmma_f32_16x16x32_f16(false, a, false, b, (short)0, c, false, false);
}

// ---------------------------------------------------------------- converts --
__global__ __launch_bounds__(256)
void cvt_f32_f16_kernel(const float* __restrict__ in, h16* __restrict__ out, int n) {
  int i = blockIdx.x * blockDim.x + threadIdx.x;
  if (i < n) out[i] = (h16)in[i];
}

// in: [rows, cols] f32 row-major -> out: [cols, rows] f16
__global__ __launch_bounds__(256)
void transpose_f32_f16_kernel(const float* __restrict__ in, h16* __restrict__ out,
                              int rows, int cols) {
  int r = blockIdx.x * blockDim.x + threadIdx.x;  // contiguous in out
  int c = blockIdx.y;
  if (r < rows) out[(size_t)c * rows + r] = (h16)in[(size_t)r * cols + c];
}

// ---------------------------------------------------------------- QKV GEMM --
// Y[m,d] = sum_k X[m,k]*Wqkv[d,k]; Wt pre-transposed [K, 3C].
// Block: 8 waves, tile 256(M) x 64(N); wave tile 32x64 (8 WMMAs / k-step).
// B tile staged in LDS via async copy, pointer-swapped double buffer.
#define GW 8
__global__ __launch_bounds__(32 * GW)
void qkv_gemm_kernel(const h16* __restrict__ X, const h16* __restrict__ Wt,
                     h16* __restrict__ Q, h16* __restrict__ Kt, h16* __restrict__ V) {
  __shared__ h16 bbuf[2][32][72];   // [k][n], padded rows vs bank conflicts
  const int tid  = threadIdx.x;
  const int lane = tid & 31;
  const int wave = tid >> 5;
  const int n0 = blockIdx.x * 64;
  const int m0 = blockIdx.y * (GW * 32) + wave * 32;

  // cooperative fill mapping: 256 threads x 16B = 4KB = 32x64 halves
  const int frow = tid >> 3;        // 0..31 (k)
  const int fcol = (tid & 7) * 8;   // 0..56 (n)
  const h16* gsrc = Wt + (size_t)frow * K3_ + n0 + fcol;

  h16* bcur = &bbuf[0][0][0];
  h16* bnxt = &bbuf[1][0][0];
  const int foff = frow * 72 + fcol;

  async_ld16(bcur + foff, gsrc);    // prologue: k0 = 0

  v8f acc[2][4] = {};
  const h16* arow = X + (size_t)m0 * C_;

#pragma clang loop unroll(disable)
  for (int i = 0; i < C_ / 32; ++i) {
    const int k0 = i * 32;
    wait_async0();        // our async stores for current buf are in LDS
    __syncthreads();      // everyone's are; all reads of next buf finished
    if (i + 1 < C_ / 32)
      async_ld16(bnxt + foff, gsrc + (size_t)(k0 + 32) * K3_);

    __builtin_prefetch(arow + k0 + 256, 0, 1);
    v16h a0 = load_afrag(arow + k0, C_, lane);
    v16h a1 = load_afrag(arow + (size_t)16 * C_ + k0, C_, lane);
    const h16* brow = bcur + (lane & 15);   // B-frag: lane&15 = n column
    // load all 4 B fragments first (distinct tuples -> one DS clause),
    // then run the 8-WMMA burst uninterrupted
    v16h b0 = ldsb(brow +  0, 72, lane);
    v16h b1 = ldsb(brow + 16, 72, lane);
    v16h b2 = ldsb(brow + 32, 72, lane);
    v16h b3 = ldsb(brow + 48, 72, lane);
    acc[0][0] = wmma_f16(a0, b0, acc[0][0]);
    acc[1][0] = wmma_f16(a1, b0, acc[1][0]);
    acc[0][1] = wmma_f16(a0, b1, acc[0][1]);
    acc[1][1] = wmma_f16(a1, b1, acc[1][1]);
    acc[0][2] = wmma_f16(a0, b2, acc[0][2]);
    acc[1][2] = wmma_f16(a1, b2, acc[1][2]);
    acc[0][3] = wmma_f16(a0, b3, acc[0][3]);
    acc[1][3] = wmma_f16(a1, b3, acc[1][3]);
    h16* tmp = bcur; bcur = bnxt; bnxt = tmp;
  }

  // scatter: Q (scaled) [B,H,N,HD], K transposed [B,H,HD,N], V [B,H,N,HD]
  const int which = n0 >> 10;
  const int c0    = n0 & (C_ - 1);
  const int h     = c0 >> 6;
  const int hd0   = c0 & (HD_ - 1);
#pragma unroll
  for (int mt = 0; mt < 2; ++mt) {
    const int mrow = m0 + mt * 16 + ((lane >> 4) << 3);
#pragma unroll
    for (int t = 0; t < 4; ++t) {
      const int hd = hd0 + t * 16 + (lane & 15);
#pragma unroll
      for (int r = 0; r < 8; ++r) {
        const int mm = mrow + r;
        const int b  = mm >> 11;
        const int n  = mm & (N_ - 1);
        const float val = acc[mt][t][r];
        const size_t bh = (size_t)(b * H_ + h);
        if (which == 0)      Q [(bh * N_ + n) * HD_ + hd] = (h16)(val * 0.125f);
        else if (which == 1) Kt[(bh * HD_ + hd) * N_ + n] = (h16)val;
        else                 V [(bh * N_ + n) * HD_ + hd] = (h16)val;
      }
    }
  }
}

// ---------------------------------------------------------- flash attention --
// 4 waves / block, each owns 16 query rows; K^T and V 32-key tiles staged in
// LDS via async copy (shared by all 4 waves), pointer-swapped double buffer.
#define AW 4
__global__ __launch_bounds__(32 * AW)
void flash_attn_kernel(const h16* __restrict__ Q, const h16* __restrict__ Kt,
                       const h16* __restrict__ V, h16* __restrict__ O) {
  __shared__ h16 kbuf[2][64][40];   // [hd][key]
  __shared__ h16 vbuf[2][32][72];   // [key][hd]
  __shared__ h16 pbuf[AW][16][40];  // per-wave P staging
  const int tid  = threadIdx.x;
  const int lane = tid & 31;
  const int wave = tid >> 5;
  const int h  = blockIdx.y;
  const int b  = blockIdx.z;
  const int m0 = blockIdx.x * (16 * AW) + wave * 16;
  const size_t bh = (size_t)(b * H_ + h);
  const h16* qp = Q  + bh * (size_t)N_ * HD_;
  const h16* kp = Kt + bh * (size_t)HD_ * N_;
  const h16* vp = V  + bh * (size_t)N_ * HD_;

  // cooperative fill: 128 threads x 2 chunks x 16B per tile
  const int i0 = tid << 1, i1 = i0 | 1;
  const int kr0 = i0 >> 2, kc0 = (i0 & 3) << 3, kr1 = i1 >> 2, kc1 = (i1 & 3) << 3;
  const int vr0 = i0 >> 3, vc0 = (i0 & 7) << 3, vr1 = i1 >> 3, vc1 = (i1 & 7) << 3;
  const int kfo0 = kr0 * 40 + kc0, kfo1 = kr1 * 40 + kc1;
  const int vfo0 = vr0 * 72 + vc0, vfo1 = vr1 * 72 + vc1;

  h16* kcur = &kbuf[0][0][0]; h16* knxt = &kbuf[1][0][0];
  h16* vcur = &vbuf[0][0][0]; h16* vnxt = &vbuf[1][0][0];

  // prologue: j0 = 0
  async_ld16(kcur + kfo0, kp + (size_t)kr0 * N_ + kc0);
  async_ld16(kcur + kfo1, kp + (size_t)kr1 * N_ + kc1);
  async_ld16(vcur + vfo0, vp + (size_t)vr0 * HD_ + vc0);
  async_ld16(vcur + vfo1, vp + (size_t)vr1 * HD_ + vc1);

  const v16h qa0 = load_afrag(qp + (size_t)m0 * HD_ + 0,  HD_, lane);
  const v16h qa1 = load_afrag(qp + (size_t)m0 * HD_ + 32, HD_, lane);

  v8f o0 = {}, o1 = {}, o2 = {}, o3 = {};
  v8f mi, li;
#pragma unroll
  for (int r = 0; r < 8; ++r) { mi[r] = -3.0e38f; li[r] = 0.0f; }

#pragma clang loop unroll(disable)
  for (int jb = 0; jb < N_ / 32; ++jb) {
    wait_async0();
    __syncthreads();
    if (jb + 1 < N_ / 32) {
      const int jn = (jb + 1) * 32;
      async_ld16(knxt + kfo0, kp + (size_t)kr0 * N_ + jn + kc0);
      async_ld16(knxt + kfo1, kp + (size_t)kr1 * N_ + jn + kc1);
      async_ld16(vnxt + vfo0, vp + (size_t)(jn + vr0) * HD_ + vc0);
      async_ld16(vnxt + vfo1, vp + (size_t)(jn + vr1) * HD_ + vc1);
    }

    // S = Q K^T : two 16x16 f32 frags over HD=64; load 4 K-frags first
    v8f s0 = {}, s1 = {};
    {
      const h16* kb0 = kcur + (lane & 15);              // hd rows 0..31
      const h16* kb1 = kcur + 32 * 40 + (lane & 15);    // hd rows 32..63
      v16h f00 = ldsb(kb0, 40, lane);
      v16h f01 = ldsb(kb0 + 16, 40, lane);
      v16h f10 = ldsb(kb1, 40, lane);
      v16h f11 = ldsb(kb1 + 16, 40, lane);
      s0 = wmma_f16(qa0, f00, s0);
      s1 = wmma_f16(qa0, f01, s1);
      s0 = wmma_f16(qa1, f10, s0);
      s1 = wmma_f16(qa1, f11, s1);
    }

    // online softmax; D-layout: VGPR r = row (r + 8*(lane>=16)), N = lane&15
    v8f p0, p1;
#pragma unroll
    for (int r = 0; r < 8; ++r) {
      float t = fmaxf(s0[r], s1[r]);
#pragma unroll
      for (int m = 1; m < 16; m <<= 1) t = fmaxf(t, __shfl_xor(t, m, 32));
      const float mn   = fmaxf(mi[r], t);
      const float corr = __expf(mi[r] - mn);
      const float a0   = __expf(s0[r] - mn);
      const float a1   = __expf(s1[r] - mn);
      float rs = a0 + a1;
#pragma unroll
      for (int m = 1; m < 16; m <<= 1) rs += __shfl_xor(rs, m, 32);
      li[r] = li[r] * corr + rs;
      mi[r] = mn;
      p0[r] = a0; p1[r] = a1;
      o0[r] *= corr; o1[r] *= corr; o2[r] *= corr; o3[r] *= corr;
    }

    // P: D-layout regs -> row-major 16x32 in wave-private LDS
    const int prow = (lane >> 4) << 3;
    const int pcol = lane & 15;
#pragma unroll
    for (int r = 0; r < 8; ++r) {
      pbuf[wave][prow + r][pcol]      = (h16)p0[r];
      pbuf[wave][prow + r][pcol + 16] = (h16)p1[r];
    }
    asm volatile("s_wait_dscnt 0" ::: "memory");  // same-wave DS in-order

    // re-read P in A-fragment layout; load 4 V-frags into distinct tuples
    v16h pa;
    {
      const int row  = lane & 15;
      const int koff = (lane >> 4) << 3;
      const h16* p = &pbuf[wave][row][koff];
      pa = cat8(*(const v8h*)p, *(const v8h*)(p + 16));
    }
    const h16* vrow = vcur + (lane & 15);
    v16h vb0 = ldsb(vrow +  0, 72, lane);
    v16h vb1 = ldsb(vrow + 16, 72, lane);
    v16h vb2 = ldsb(vrow + 32, 72, lane);
    v16h vb3 = ldsb(vrow + 48, 72, lane);
    o0 = wmma_f16(pa, vb0, o0);
    o1 = wmma_f16(pa, vb1, o1);
    o2 = wmma_f16(pa, vb2, o2);
    o3 = wmma_f16(pa, vb3, o3);

    h16* t0 = kcur; kcur = knxt; knxt = t0;
    h16* t1 = vcur; vcur = vnxt; vnxt = t1;
  }

  // normalize, store O as f16 [B, N, C]
  v8f inv;
#pragma unroll
  for (int r = 0; r < 8; ++r) inv[r] = 1.0f / li[r];
  h16* op = O + ((size_t)b * N_ + m0) * C_ + h * HD_;
  const int rowoff = (lane >> 4) << 3;
  const int nn = lane & 15;
#pragma unroll
  for (int r = 0; r < 8; ++r) {
    const size_t rb = (size_t)(rowoff + r) * C_;
    op[rb +  0 + nn] = (h16)(o0[r] * inv[r]);
    op[rb + 16 + nn] = (h16)(o1[r] * inv[r]);
    op[rb + 32 + nn] = (h16)(o2[r] * inv[r]);
    op[rb + 48 + nn] = (h16)(o3[r] * inv[r]);
  }
}

// ----------------------------------------------------------- out projection --
__global__ __launch_bounds__(32 * GW)
void out_proj_kernel(const h16* __restrict__ A, const h16* __restrict__ Wt,
                     const float* __restrict__ bias, float* __restrict__ Y) {
  __shared__ h16 bbuf[2][32][72];
  const int tid  = threadIdx.x;
  const int lane = tid & 31;
  const int wave = tid >> 5;
  const int n0 = blockIdx.x * 64;
  const int m0 = blockIdx.y * (GW * 32) + wave * 32;

  const int frow = tid >> 3;
  const int fcol = (tid & 7) * 8;
  const h16* gsrc = Wt + (size_t)frow * C_ + n0 + fcol;

  h16* bcur = &bbuf[0][0][0];
  h16* bnxt = &bbuf[1][0][0];
  const int foff = frow * 72 + fcol;

  async_ld16(bcur + foff, gsrc);

  v8f acc[2][4] = {};
  const h16* arow = A + (size_t)m0 * C_;

#pragma clang loop unroll(disable)
  for (int i = 0; i < C_ / 32; ++i) {
    const int k0 = i * 32;
    wait_async0();
    __syncthreads();
    if (i + 1 < C_ / 32)
      async_ld16(bnxt + foff, gsrc + (size_t)(k0 + 32) * C_);

    __builtin_prefetch(arow + k0 + 256, 0, 1);
    v16h a0 = load_afrag(arow + k0, C_, lane);
    v16h a1 = load_afrag(arow + (size_t)16 * C_ + k0, C_, lane);
    const h16* brow = bcur + (lane & 15);
    v16h b0 = ldsb(brow +  0, 72, lane);
    v16h b1 = ldsb(brow + 16, 72, lane);
    v16h b2 = ldsb(brow + 32, 72, lane);
    v16h b3 = ldsb(brow + 48, 72, lane);
    acc[0][0] = wmma_f16(a0, b0, acc[0][0]);
    acc[1][0] = wmma_f16(a1, b0, acc[1][0]);
    acc[0][1] = wmma_f16(a0, b1, acc[0][1]);
    acc[1][1] = wmma_f16(a1, b1, acc[1][1]);
    acc[0][2] = wmma_f16(a0, b2, acc[0][2]);
    acc[1][2] = wmma_f16(a1, b2, acc[1][2]);
    acc[0][3] = wmma_f16(a0, b3, acc[0][3]);
    acc[1][3] = wmma_f16(a1, b3, acc[1][3]);
    h16* tmp = bcur; bcur = bnxt; bnxt = tmp;
  }

#pragma unroll
  for (int mt = 0; mt < 2; ++mt) {
    const int mrow = m0 + mt * 16 + ((lane >> 4) << 3);
#pragma unroll
    for (int t = 0; t < 4; ++t) {
      const int nn = n0 + t * 16 + (lane & 15);
      const float bv = bias[nn];
#pragma unroll
      for (int r = 0; r < 8; ++r)
        Y[(size_t)(mrow + r) * C_ + nn] = acc[mt][t][r] + bv;
    }
  }
}

// ------------------------------------------------------------------- launch --
extern "C" void kernel_launch(void* const* d_in, const int* in_sizes, int n_in,
                              void* d_out, int out_size, void* d_ws, size_t ws_size,
                              hipStream_t stream) {
  const float* x     = (const float*)d_in[0];
  const float* w_qkv = (const float*)d_in[1];
  const float* w_out = (const float*)d_in[2];
  const float* b_out = (const float*)d_in[3];
  float* out = (float*)d_out;

  char* ws = (char*)d_ws;
  size_t off = 0;
  auto take = [&](size_t bytes) -> char* {
    char* p = ws + off;
    off += (bytes + 255) & ~(size_t)255;
    return p;
  };
  h16* xh    = (h16*)take((size_t)M_ * C_ * 2);
  h16* wqkvT = (h16*)take((size_t)C_ * K3_ * 2);
  h16* woutT = (h16*)take((size_t)C_ * C_ * 2);
  h16* Q     = (h16*)take((size_t)M_ * C_ * 2);
  h16* Kt    = (h16*)take((size_t)M_ * C_ * 2);
  h16* V     = (h16*)take((size_t)M_ * C_ * 2);
  h16* O     = (h16*)take((size_t)M_ * C_ * 2);

  cvt_f32_f16_kernel<<<(M_ * C_ + 255) / 256, 256, 0, stream>>>(x, xh, M_ * C_);
  transpose_f32_f16_kernel<<<dim3((K3_ + 255) / 256, C_), 256, 0, stream>>>(w_qkv, wqkvT, K3_, C_);
  transpose_f32_f16_kernel<<<dim3((C_ + 255) / 256, C_), 256, 0, stream>>>(w_out, woutT, C_, C_);

  qkv_gemm_kernel<<<dim3(K3_ / 64, M_ / (GW * 32)), 32 * GW, 0, stream>>>(xh, wqkvT, Q, Kt, V);
  flash_attn_kernel<<<dim3(N_ / (16 * AW), H_, B_), 32 * AW, 0, stream>>>(Q, Kt, V, O);
  out_proj_kernel<<<dim3(C_ / 64, M_ / (GW * 32)), 32 * GW, 0, stream>>>(O, woutT, b_out, out);
}
